// AttentionStoreProcessor_26929444946238
// MI455X (gfx1250) — matrix operand order repaired and checked by reading, not compiled, not run on hardware
//
#include <hip/hip_runtime.h>
#include <hip/hip_bf16.h>

typedef _Float16 f16;
typedef __attribute__((ext_vector_type(16))) _Float16 v16h;
typedef __attribute__((ext_vector_type(8)))  _Float16 v8h;
typedef __attribute__((ext_vector_type(8)))  float    v8f;

#define S_TOK  2048
#define DMODEL 1280
#define NHEAD  20
#define HDIM   64
// hd^-0.5 * log2(e): scores land in log2 domain for raw v_exp_f32 softmax
#define QSCALE 0.1803368801111204f

static __device__ __forceinline__ v16h cat16(v8h lo, v8h hi) {
  return __builtin_shufflevector(lo, hi, 0,1,2,3,4,5,6,7,8,9,10,11,12,13,14,15);
}
static __device__ __forceinline__ v8f wmma16(v16h a, v16h b, v8f c) {
  return __builtin_amdgcn_wmma_f32_16x16x32_f16(false, a, false, b, (short)0, c, false, false);
}
static __device__ __forceinline__ v8f zero8() {
  v8f z = {0.f,0.f,0.f,0.f,0.f,0.f,0.f,0.f};
  return z;
}

// VALU cross-lane via DPP16 ROW_XMASK (0x160|mask): butterfly within 16-lane halves.
template<int CTRL>
static __device__ __forceinline__ float dppf(float v) {
  return __int_as_float(__builtin_amdgcn_update_dpp(0, __float_as_int(v), CTRL, 0xf, 0xf, true));
}
static __device__ __forceinline__ float red_max16(float x) {
  x = fmaxf(x, dppf<0x161>(x));
  x = fmaxf(x, dppf<0x162>(x));
  x = fmaxf(x, dppf<0x164>(x));
  x = fmaxf(x, dppf<0x168>(x));
  return x;
}
static __device__ __forceinline__ float red_sum16(float x) {
  x += dppf<0x161>(x);
  x += dppf<0x162>(x);
  x += dppf<0x164>(x);
  x += dppf<0x168>(x);
  return x;
}

// CDNA5 async memory->LDS copy, 16 bytes per lane (tracked by ASYNCcnt, in-order per wave).
static __device__ __forceinline__ void async_cp16(const f16* lds_dst, const f16* gsrc) {
  unsigned lds_off = (unsigned)(unsigned long long)lds_dst;   // LDS aperture: addr[31:0] = LDS byte addr
  unsigned long long ga = (unsigned long long)gsrc;
  asm volatile("global_load_async_to_lds_b128 %0, %1, off"
               :: "v"(lds_off), "v"(ga) : "memory");
}
static __device__ __forceinline__ void wait_async0()  { asm volatile("s_wait_asynccnt 0x0"  ::: "memory"); }
static __device__ __forceinline__ void wait_async4()  { asm volatile("s_wait_asynccnt 0x4"  ::: "memory"); }
static __device__ __forceinline__ void wait_async8()  { asm volatile("s_wait_asynccnt 0x8"  ::: "memory"); }

// ---------------------------------------------------------------- convert f32->f16
__global__ void k_f32_to_f16(const float* __restrict__ x, f16* __restrict__ y, int n) {
  int i = blockIdx.x * 256 + threadIdx.x;
  if (i < n) y[i] = (f16)x[i];
}

// ------------------------- transpose + convert: 4 weights in one launch (z selects)
__global__ void k_transpose_f16(const float* __restrict__ W0, const float* __restrict__ W1,
                                const float* __restrict__ W2, const float* __restrict__ W3,
                                f16* __restrict__ T0, f16* __restrict__ T1,
                                f16* __restrict__ T2, f16* __restrict__ T3) {
  const float* W = (blockIdx.z == 0) ? W0 : (blockIdx.z == 1) ? W1 : (blockIdx.z == 2) ? W2 : W3;
  f16*        WT = (blockIdx.z == 0) ? T0 : (blockIdx.z == 1) ? T1 : (blockIdx.z == 2) ? T2 : T3;
  __shared__ float t[32][33];
  int tx = threadIdx.x, ty = threadIdx.y;
  int n = blockIdx.x * 32 + tx;
  #pragma unroll
  for (int j = 0; j < 32; j += 8) {
    int k = blockIdx.y * 32 + ty + j;
    t[ty + j][tx] = W[(size_t)k * DMODEL + n];
  }
  __syncthreads();
  int k2 = blockIdx.y * 32 + tx;
  #pragma unroll
  for (int j = 0; j < 32; j += 8) {
    int n2 = blockIdx.x * 32 + ty + j;
    WT[(size_t)n2 * DMODEL + k2] = (f16)t[tx][ty + j];
  }
}

// ---------------------------------------------------------------- tiled WMMA GEMM
// C[M,N] (f32) = A[M,K] (f16 row-major) * BT[N,K]^T.  blockIdx.z selects (BT, C) triple.
// 4-deep async-to-LDS ring: one barrier per K-stage, 2 chunks in flight.
template<bool EPI>
__global__ __launch_bounds__(256) void k_gemm_f16(
    const f16* __restrict__ A,
    const f16* __restrict__ B0, const f16* __restrict__ B1, const f16* __restrict__ B2,
    float* __restrict__ C0, float* __restrict__ C1, float* __restrict__ C2,
    int M, int N, int K,
    const float* __restrict__ bias, const float* __restrict__ residual)
{
  const f16* BT = (blockIdx.z == 0) ? B0 : (blockIdx.z == 1) ? B1 : B2;
  float*     C  = (blockIdx.z == 0) ? C0 : (blockIdx.z == 1) ? C1 : C2;

  __shared__ f16 lA[4][128 * 40];   // 80B row pitch, 16B aligned, conflict-free
  __shared__ f16 lB[4][128 * 40];
  const int tid  = threadIdx.x;
  const int wave = tid >> 5, lane = tid & 31;
  const int l15  = lane & 15, lhi = lane >> 4;
  const int wm   = wave >> 1, wn = wave & 1;       // 4x2 wave grid -> 32x64 per wave
  const int m0   = blockIdx.y * 128, n0 = blockIdx.x * 128;

  v8f acc[2][4];
  #pragma unroll
  for (int i = 0; i < 2; ++i)
    #pragma unroll
    for (int j = 0; j < 4; ++j) acc[i][j] = zero8();

  const int lr = tid >> 2;          // 0..63
  const int lc = (tid & 3) * 8;     // 0,8,16,24

  auto issue = [&](int pb, int kb) {   // 4 async ops per thread per chunk
    #pragma unroll
    for (int j = 0; j < 2; ++j) {
      int row = lr + j * 64;
      async_cp16(&lA[pb][row * 40 + lc], &A [(size_t)(m0 + row) * K + kb + lc]);
      async_cp16(&lB[pb][row * 40 + lc], &BT[(size_t)(n0 + row) * K + kb + lc]);
    }
  };

  const int niter = K >> 5;          // K / 32
  issue(0, 0);
  if (niter > 1) issue(1, 32);
  for (int i = 0; i < niter; ++i) {
    if (i + 2 < niter) { issue((i + 2) & 3, (i + 2) * 32); wait_async8(); }
    else if (i + 1 < niter) wait_async4();
    else wait_async0();
    __syncthreads();                 // chunk i visible; ring depth 4 makes this sufficient

    const int pb = i & 3;
    v16h af[2], bf[4];
    #pragma unroll
    for (int mt = 0; mt < 2; ++mt) {
      const f16* p = &lA[pb][(wm * 32 + mt * 16 + l15) * 40 + lhi * 8];
      af[mt] = cat16(*(const v8h*)p, *(const v8h*)(p + 16));
    }
    #pragma unroll
    for (int nt = 0; nt < 4; ++nt) {
      const f16* p = &lB[pb][(wn * 64 + nt * 16 + l15) * 40 + lhi * 8];
      bf[nt] = cat16(*(const v8h*)p, *(const v8h*)(p + 16));
    }
    #pragma unroll
    for (int mt = 0; mt < 2; ++mt)
      #pragma unroll
      for (int nt = 0; nt < 4; ++nt)
        acc[mt][nt] = wmma16(af[mt], bf[nt], acc[mt][nt]);
  }

  const int ncol0 = n0 + wn * 64 + l15;
  #pragma unroll
  for (int mt = 0; mt < 2; ++mt)
    #pragma unroll
    for (int r = 0; r < 8; ++r) {
      const int R = m0 + wm * 32 + mt * 16 + lhi * 8 + r;
      float*       crow = C + (size_t)R * N + ncol0;
      const float* rrow = EPI ? residual + (size_t)R * N + ncol0 : nullptr;
      #pragma unroll
      for (int nt = 0; nt < 4; ++nt) {
        float v = acc[mt][nt][r];
        if (EPI) v += bias[ncol0 + nt * 16] + rrow[nt * 16];
        crow[nt * 16] = v;
      }
    }
}

// ------------------------------- CAPE rotation + layout: Q,K -> [h][s][64], V -> [h][64][s]
// Q pre-scaled by hd^-0.5 * log2(e) -> softmax runs on raw exp2.
__global__ void k_prep(const float* __restrict__ qraw, const float* __restrict__ kraw,
                       const float* __restrict__ vraw,
                       const float* __restrict__ p_out, const float* __restrict__ p_inv,
                       f16* __restrict__ Q, f16* __restrict__ Kt, f16* __restrict__ V)
{
  int gid = blockIdx.x * 256 + threadIdx.x;
  if (gid >= S_TOK * (DMODEL / 4)) return;
  int s = gid / (DMODEL / 4);
  int c = (gid % (DMODEL / 4)) * 4;
  int t = s >> 10;                        // frame index (l = 1024)
  const float* Pq = p_inv + t * 16;       // q uses p_out_inv
  const float* Pk = p_out + t * 16;       // k uses p_out
  float q[4], k[4], v[4];
  #pragma unroll
  for (int i = 0; i < 4; ++i) {
    q[i] = qraw[(size_t)s * DMODEL + c + i];
    k[i] = kraw[(size_t)s * DMODEL + c + i];
    v[i] = vraw[(size_t)s * DMODEL + c + i];
  }
  int h = c >> 6, hd = c & 63;
  size_t qb = ((size_t)h * S_TOK + s) * HDIM + hd;
  #pragma unroll
  for (int j = 0; j < 4; ++j) {
    float qq = q[0]*Pq[j] + q[1]*Pq[4+j] + q[2]*Pq[8+j] + q[3]*Pq[12+j];
    float kk = k[0]*Pk[j] + k[1]*Pk[4+j] + k[2]*Pk[8+j] + k[3]*Pk[12+j];
    Q [qb + j] = (f16)(qq * QSCALE);
    Kt[qb + j] = (f16)kk;
    V[((size_t)h * HDIM + hd + j) * S_TOK + s] = (f16)v[j];
  }
}

// ---------------------------------------------------------------- flash attention
// grid (16 q-blocks, 20 heads), 256 threads = 8 waves, each wave: 16 q-rows x 64 hd.
// 128 q-rows share each K/V chunk (halves L2 traffic). 4-deep async-to-LDS ring.
__global__ __launch_bounds__(256) void k_attn(
    const f16* __restrict__ Q, const f16* __restrict__ Kk, const f16* __restrict__ V,
    f16* __restrict__ O)
{
  __shared__ f16 lK[4][64 * 72];      // [sk][hd]  (B operand for S)
  __shared__ f16 lV[4][64 * 72];      // [hd][sk]  (B operand for O)
  __shared__ f16 lP[8][16 * 72];      // per-wave P tile in A-operand layout
  const int tid = threadIdx.x;
  const int wave = tid >> 5, lane = tid & 31;
  const int l15 = lane & 15, lhi = lane >> 4;
  const int h  = blockIdx.y;
  const int qb = blockIdx.x * 128;
  const f16* Qh = Q  + (size_t)h * S_TOK * HDIM;
  const f16* Kh = Kk + (size_t)h * S_TOK * HDIM;
  const f16* Vh = V  + (size_t)h * HDIM * S_TOK;

  // Q fragments live in registers for the whole pass
  v16h aq[2];
  {
    int row = qb + wave * 16 + l15;
    #pragma unroll
    for (int kc = 0; kc < 2; ++kc) {
      const f16* p = Qh + (size_t)row * HDIM + kc * 32 + lhi * 8;
      aq[kc] = cat16(*(const v8h*)p, *(const v8h*)(p + 16));
    }
  }

  v8f o[4];
  #pragma unroll
  for (int nt = 0; nt < 4; ++nt) o[nt] = zero8();
  float mrun[8], lrun[8];
  #pragma unroll
  for (int r = 0; r < 8; ++r) { mrun[r] = -3.0e38f; lrun[r] = 0.f; }

  const int lr  = tid >> 2;          // 0..63
  const int lc4 = (tid & 3) * 8;     // 0,8,16,24

  auto issue = [&](int pb, int sk0) {   // 4 async ops per thread per chunk
    async_cp16(&lK[pb][lr * 72 + lc4],      &Kh[(size_t)(sk0 + lr) * HDIM + lc4]);
    async_cp16(&lK[pb][lr * 72 + lc4 + 32], &Kh[(size_t)(sk0 + lr) * HDIM + lc4 + 32]);
    async_cp16(&lV[pb][lr * 72 + lc4],      &Vh[(size_t)lr * S_TOK + sk0 + lc4]);
    async_cp16(&lV[pb][lr * 72 + lc4 + 32], &Vh[(size_t)lr * S_TOK + sk0 + lc4 + 32]);
  };

  const int niter = S_TOK / 64;     // 32
  issue(0, 0);
  issue(1, 64);
  for (int i = 0; i < niter; ++i) {
    if (i + 2 < niter) { issue((i + 2) & 3, (i + 2) * 64); wait_async8(); }
    else if (i + 1 < niter) wait_async4();
    else wait_async0();
    __syncthreads();                 // chunk i visible; ring depth 4 -> no trailing barrier

    const int pb = i & 3;
    // S = Q * K^T  (16 x 64, K=64); Q pre-scaled, scores in log2 domain
    v8f s[4];
    #pragma unroll
    for (int nt = 0; nt < 4; ++nt) s[nt] = zero8();
    #pragma unroll
    for (int kc = 0; kc < 2; ++kc) {
      #pragma unroll
      for (int nt = 0; nt < 4; ++nt) {
        const f16* p = &lK[pb][(nt * 16 + l15) * 72 + kc * 32 + lhi * 8];
        v16h bf = cat16(*(const v8h*)p, *(const v8h*)(p + 16));
        s[nt] = wmma16(aq[kc], bf, s[nt]);
      }
    }

    // online softmax (row = lhi*8 + r, spread across lanes 0..15 of each half)
    #pragma unroll
    for (int r = 0; r < 8; ++r) {
      float mx = red_max16(fmaxf(fmaxf(s[0][r], s[1][r]), fmaxf(s[2][r], s[3][r])));
      float nm = fmaxf(mrun[r], mx);
      float p0 = __builtin_amdgcn_exp2f(s[0][r] - nm);
      float p1 = __builtin_amdgcn_exp2f(s[1][r] - nm);
      float p2 = __builtin_amdgcn_exp2f(s[2][r] - nm);
      float p3 = __builtin_amdgcn_exp2f(s[3][r] - nm);
      float rs = red_sum16((p0 + p1) + (p2 + p3));
      float corr = __builtin_amdgcn_exp2f(mrun[r] - nm);
      mrun[r] = nm;
      lrun[r] = lrun[r] * corr + rs;
      #pragma unroll
      for (int nt = 0; nt < 4; ++nt) o[nt][r] *= corr;
      int prow = lhi * 8 + r;
      lP[wave][prow * 72 +  0 + l15] = (f16)p0;
      lP[wave][prow * 72 + 16 + l15] = (f16)p1;
      lP[wave][prow * 72 + 32 + l15] = (f16)p2;
      lP[wave][prow * 72 + 48 + l15] = (f16)p3;
    }
    // lP is wave-private: same-wave DS ordering covers the RAW, no barrier needed.

    // O += P * V  (16 x 64, K=64)
    #pragma unroll
    for (int kc = 0; kc < 2; ++kc) {
      const f16* pp = &lP[wave][l15 * 72 + kc * 32 + lhi * 8];
      v16h ap = cat16(*(const v8h*)pp, *(const v8h*)(pp + 16));
      #pragma unroll
      for (int nt = 0; nt < 4; ++nt) {
        const f16* pv = &lV[pb][(nt * 16 + l15) * 72 + kc * 32 + lhi * 8];
        v16h bv = cat16(*(const v8h*)pv, *(const v8h*)(pv + 16));
        o[nt] = wmma16(ap, bv, o[nt]);
      }
    }
  }

  // normalize + store f16 [s][d]
  #pragma unroll
  for (int r = 0; r < 8; ++r) {
    float inv = 1.0f / lrun[r];
    int row = qb + wave * 16 + lhi * 8 + r;
    #pragma unroll
    for (int nt = 0; nt < 4; ++nt)
      O[(size_t)row * DMODEL + h * HDIM + nt * 16 + l15] = (f16)(o[nt][r] * inv);
  }
}

// ---------------------------------------------------------------- host launcher
static inline void* carve(char*& p, size_t bytes) {
  void* r = (void*)p;
  p += (bytes + 255) & ~(size_t)255;
  return r;
}

extern "C" void kernel_launch(void* const* d_in, const int* in_sizes, int n_in,
                              void* d_out, int out_size, void* d_ws, size_t ws_size,
                              hipStream_t stream) {
  (void)in_sizes; (void)n_in; (void)out_size; (void)ws_size;
  const float* hs   = (const float*)d_in[0];
  const float* pout = (const float*)d_in[1];
  const float* pinv = (const float*)d_in[2];
  const float* Wq   = (const float*)d_in[3];
  const float* Wk   = (const float*)d_in[4];
  const float* Wv   = (const float*)d_in[5];
  const float* Wo   = (const float*)d_in[6];
  const float* bo   = (const float*)d_in[7];
  float* out = (float*)d_out;

  char* p = (char*)d_ws;
  f16*   hs16  = (f16*)  carve(p, (size_t)S_TOK * DMODEL * 2);
  f16*   wqT   = (f16*)  carve(p, (size_t)DMODEL * DMODEL * 2);
  f16*   wkT   = (f16*)  carve(p, (size_t)DMODEL * DMODEL * 2);
  f16*   wvT   = (f16*)  carve(p, (size_t)DMODEL * DMODEL * 2);
  f16*   woT   = (f16*)  carve(p, (size_t)DMODEL * DMODEL * 2);
  float* qraw  = (float*)carve(p, (size_t)S_TOK * DMODEL * 4);
  float* kraw  = (float*)carve(p, (size_t)S_TOK * DMODEL * 4);
  float* vraw  = (float*)carve(p, (size_t)S_TOK * DMODEL * 4);
  f16*   Qe    = (f16*)  carve(p, (size_t)S_TOK * DMODEL * 2);
  f16*   Ke    = (f16*)  carve(p, (size_t)S_TOK * DMODEL * 2);
  f16*   Ve    = (f16*)  carve(p, (size_t)S_TOK * DMODEL * 2);
  f16*   att16 = (f16*)  carve(p, (size_t)S_TOK * DMODEL * 2);

  // 1. convert activations
  k_f32_to_f16<<<(S_TOK * DMODEL + 255) / 256, 256, 0, stream>>>(hs, hs16, S_TOK * DMODEL);

  // 2. transpose+convert all 4 weights -> [N][K] f16 (one launch, z selects)
  k_transpose_f16<<<dim3(DMODEL / 32, DMODEL / 32, 4), dim3(32, 8), 0, stream>>>(
      Wq, Wk, Wv, Wo, wqT, wkT, wvT, woT);

  // 3. QKV projections in one launch (z selects weight/output)
  dim3 gg3(DMODEL / 128, S_TOK / 128, 3);
  k_gemm_f16<false><<<gg3, 256, 0, stream>>>(hs16, wqT, wkT, wvT, qraw, kraw, vraw,
                                             S_TOK, DMODEL, DMODEL, nullptr, nullptr);

  // 4. CAPE + attention layouts (Q pre-scaled by hd^-0.5 * log2e)
  k_prep<<<(S_TOK * (DMODEL / 4) + 255) / 256, 256, 0, stream>>>(
      qraw, kraw, vraw, pout, pinv, Qe, Ke, Ve);

  // 5. flash attention
  k_attn<<<dim3(S_TOK / 128, NHEAD), 256, 0, stream>>>(Qe, Ke, Ve, att16);

  // 6. output projection + bias + residual
  dim3 gg1(DMODEL / 128, S_TOK / 128, 1);
  k_gemm_f16<true><<<gg1, 256, 0, stream>>>(att16, woT, woT, woT, out, out, out,
                                            S_TOK, DMODEL, DMODEL, bo, hs);
}
